// GraphConvolution_79285096284464
// MI455X (gfx1250) — matrix operand order repaired and checked by reading, not compiled, not run on hardware
//
#include <hip/hip_runtime.h>

// ---------------------------------------------------------------------------
// Types for CDNA5 WMMA (wave32): v16bf A/B fragments, v8f accumulator.
// ---------------------------------------------------------------------------
typedef __attribute__((ext_vector_type(16))) __bf16 v16bf;
typedef __attribute__((ext_vector_type(8)))  float  v8f;
typedef __attribute__((ext_vector_type(4)))  int    v4i;

// ---- feature probes -------------------------------------------------------
#if defined(__has_builtin)
#  if __has_builtin(__builtin_amdgcn_cvt_pk_bf16_f32)
#    define HAVE_CVT_PK_BF16 1
#  endif
#  if __has_builtin(__builtin_amdgcn_global_load_async_to_lds_b128) && \
      __has_builtin(__builtin_amdgcn_s_wait_asynccnt)
#    define USE_ASYNC_LDS 1
#  endif
#endif
#ifndef HAVE_CVT_PK_BF16
#  define HAVE_CVT_PK_BF16 0
#endif
#ifndef USE_ASYNC_LDS
#  define USE_ASYNC_LDS 0
#endif

#if USE_ASYNC_LDS
typedef __attribute__((address_space(1))) v4i* gptr_v4i;
typedef __attribute__((address_space(3))) v4i* lptr_v4i;
#endif

static __device__ __forceinline__ unsigned short f2bf(float f) {
  // round-to-nearest-even float -> bf16 (bit pattern), scalar fallback
  unsigned u = __builtin_bit_cast(unsigned, f);
  u += 0x7FFFu + ((u >> 16) & 1u);
  return (unsigned short)(u >> 16);
}

#if HAVE_CVT_PK_BF16
static __device__ __forceinline__ unsigned pk_bf16(float a, float b) {
  // v_cvt_pk_bf16_f32: dst.lo = cvt(a), dst.hi = cvt(b)
  auto v = __builtin_amdgcn_cvt_pk_bf16_f32(a, b);
  return __builtin_bit_cast(unsigned, v);
}
#else
static __device__ __forceinline__ unsigned pk_bf16(float a, float b) {
  return (unsigned)f2bf(a) | ((unsigned)f2bf(b) << 16);
}
#endif

struct U128x2 { uint4 a, b; };

static __device__ __forceinline__ v16bf make_frag2(const unsigned short* p0,
                                                   const unsigned short* p1) {
  U128x2 t;
  t.a = *(const uint4*)p0;
  t.b = *(const uint4*)p1;
  return __builtin_bit_cast(v16bf, t);
}

// ---------------------------------------------------------------------------
// Helper kernels
// ---------------------------------------------------------------------------

// One wave per (r,p) row: sum over D, then rs = s>0 ? 1/sqrt(s) : 0
__global__ void __launch_bounds__(256)
row_scale_kernel(const float* __restrict__ S, float* __restrict__ rs,
                 int RP, int D) {
  int w    = blockIdx.x * 8 + (threadIdx.x >> 5);
  int lane = threadIdx.x & 31;
  if (w >= RP) return;
  size_t base = (size_t)w * D;
  float s = 0.f;
  for (int d = lane; d < D; d += 32) s += S[base + d];
#pragma unroll
  for (int off = 16; off > 0; off >>= 1) s += __shfl_xor(s, off, 32);
  if (lane == 0) rs[w] = (s > 0.f) ? rsqrtf(s) : 0.f;
}

// Partial column sums over a 128-row P slice, atomically accumulated.
__global__ void __launch_bounds__(256)
col_sum_kernel(const float* __restrict__ S, float* __restrict__ colsum,
               int P, int D) {
  int r  = blockIdx.x;
  int d  = blockIdx.y * 256 + threadIdx.x;
  int p0 = blockIdx.z * 128;
  size_t base = ((size_t)r * P + p0) * D + d;
  float s = 0.f;
#pragma unroll 4
  for (int p = 0; p < 128; ++p) s += S[base + (size_t)p * D];
  atomicAdd(&colsum[(size_t)r * D + d], s);
}

__global__ void __launch_bounds__(256)
inv_sqrt_kernel(const float* __restrict__ in, float* __restrict__ out, int n) {
  int i = blockIdx.x * 256 + threadIdx.x;
  if (i < n) { float s = in[i]; out[i] = (s > 0.f) ? rsqrtf(s) : 0.f; }
}

// Cumulative sum over r of p_weight[r][i][h], stored transposed as bf16:
// wT[r][h][i]  (so GEMM B operand is K(=i)-contiguous)
__global__ void __launch_bounds__(256)
wcum_kernel(const float* __restrict__ W, unsigned short* __restrict__ wT,
            int I, int H, int R) {
  int idx = blockIdx.x * 256 + threadIdx.x;
  if (idx >= I * H) return;
  int i = idx % I, h = idx / I;
  float acc = 0.f;
  for (int r = 0; r < R; ++r) {
    acc += W[((size_t)r * I + i) * H + h];
    wT[((size_t)r * H + h) * I + i] = f2bf(acc);
  }
}

// ---------------------------------------------------------------------------
// Tiled bf16 WMMA GEMM:  C[M,N] (+)= A[M,K] * B[N,K]^T   (B stored n-major,
// k-contiguous).  A comes from f32 memory, converted to bf16 in LDS, with
// optional transpose (A[m,k] = mem[k*M + m]) and optional per-row/col scaling
// (A[m,k] *= scaleM[r*M+m] * scaleK[r*K+k]).
//
// ACC_R=1: accumulate over r = 0..R-1, epilogue = f32 bias + LeakyReLU(0.1).
// ACC_R=0: r = blockIdx.z (A shared across r), epilogue = packed bf16
//          transposed store outT[r][n][m].
//
// Block: 256 threads = 8 wave32s (4(M) x 2(N)); block tile 128x128; wave tile
// 32x64 (2x4 fragments of 16x16); K-chunk 32.  Double-buffered LDS pipeline:
// tile s+1 is loaded (B via global_load_async_to_lds_b128 when available,
// A via scaled f32->bf16 VALU path) while WMMAs consume tile s.
// ---------------------------------------------------------------------------
#define KC  32
#define BMT 128
#define BNT 128
#define LDK 40   // padded bf16 row stride in LDS (80 B, 16B-aligned)

template<int TRANS, int SCALED, int ACC_R>
__global__ void __launch_bounds__(256)
gemm_bf16(const float* __restrict__ A,
          const unsigned short* __restrict__ B,
          const float* __restrict__ scaleM,
          const float* __restrict__ scaleK,
          const float* __restrict__ bias,
          float* __restrict__ outF,
          unsigned short* __restrict__ outT,
          int M, int N, int K, int R) {
  __shared__ __align__(16) unsigned short sA[2][BMT * LDK];
  __shared__ __align__(16) unsigned short sB[2][BNT * LDK];

  const int tid  = threadIdx.x;
  const int lane = tid & 31;
  const int hf   = lane >> 4;   // half-wave select
  const int ln   = lane & 15;
  const int wid  = tid >> 5;
  const int wM   = wid & 3;     // wave row 0..3
  const int wN   = wid >> 2;    // wave col 0..1
  const int m0   = blockIdx.x * BMT;
  const int n0   = blockIdx.y * BNT;
  const int lda  = TRANS ? M : K;   // memory row stride of the A source

  const int kSteps = K / KC;
  const int rBeg   = ACC_R ? 0 : (int)blockIdx.z;
  const int total  = (ACC_R ? R : 1) * kSteps;

  v8f c[2][4];
  const v8f vz = {0.f, 0.f, 0.f, 0.f, 0.f, 0.f, 0.f, 0.f};
#pragma unroll
  for (int i = 0; i < 2; ++i)
#pragma unroll
    for (int j = 0; j < 4; ++j) c[i][j] = vz;

  // ---- tile loader: fills sA[buf]/sB[buf] with step-s tile ----------------
  auto loadTile = [&](int s, int buf) {
    const int rr = s / kSteps;             // 0 for !ACC_R
    const int r  = rBeg + rr;
    const int k0 = (s - rr * kSteps) * KC;
    const float* Ap = A + (ACC_R ? (size_t)rr * M * K : (size_t)0);
    const unsigned short* Bp = B + (size_t)r * N * K;
    unsigned short* a = sA[buf];
    unsigned short* b = sB[buf];

    // ---- A tile: f32 -> (scale) -> bf16 into a[m][k] ----
    if (!TRANS) {
      // 128 rows x 32 f32; 8 float4 per row; 4 float4 per thread
#pragma unroll
      for (int i = 0; i < 4; ++i) {
        int idx = tid + i * 256;
        int row = idx >> 3;
        int c4  = (idx & 7) << 2;
        float4 v = *(const float4*)(Ap + (size_t)(m0 + row) * lda + k0 + c4);
        float sm = SCALED ? scaleM[(size_t)r * M + m0 + row] : 1.0f;
        float f0 = v.x * sm, f1 = v.y * sm, f2 = v.z * sm, f3 = v.w * sm;
        if (SCALED) {
          const float* skp = scaleK + (size_t)r * K + k0 + c4;
          f0 *= skp[0]; f1 *= skp[1]; f2 *= skp[2]; f3 *= skp[3];
        }
        uint2 d2;
        d2.x = pk_bf16(f0, f1);
        d2.y = pk_bf16(f2, f3);
        *(uint2*)&a[row * LDK + c4] = d2;
      }
    } else {
      // memory tile: 32 rows (k) x 128 f32 (m); store transposed to a[m][k]
#pragma unroll
      for (int i = 0; i < 4; ++i) {
        int idx = tid + i * 256;
        int kr  = idx >> 5;
        int c4  = (idx & 31) << 2;
        float4 v = *(const float4*)(Ap + (size_t)(k0 + kr) * lda + m0 + c4);
        float vv[4] = {v.x, v.y, v.z, v.w};
        float sk = SCALED ? scaleK[(size_t)r * K + k0 + kr] : 1.0f;
#pragma unroll
        for (int e = 0; e < 4; ++e) {
          float sm = SCALED ? scaleM[(size_t)r * M + m0 + c4 + e] : 1.0f;
          a[(c4 + e) * LDK + kr] = f2bf(vv[e] * sk * sm);
        }
      }
    }

    // ---- B tile: bf16 b128 copy into b[n][k] ----
#pragma unroll
    for (int i = 0; i < 2; ++i) {
      int idx = tid + i * 256;
      int row = idx >> 2;
      int cc  = (idx & 3) << 3;
      const unsigned short* src = Bp + (size_t)(n0 + row) * K + k0 + cc;
      unsigned short* dst = &b[row * LDK + cc];
#if USE_ASYNC_LDS
      __builtin_amdgcn_global_load_async_to_lds_b128(
          (gptr_v4i)(void*)src, (lptr_v4i)(void*)dst, 0, 0);
#else
      *(uint4*)dst = *(const uint4*)src;
#endif
    }
  };

  // ---- double-buffered main pipeline --------------------------------------
  loadTile(0, 0);
  for (int s = 0; s < total; ++s) {
    const int cur = s & 1;
#if USE_ASYNC_LDS
    __builtin_amdgcn_s_wait_asynccnt(0);   // own async B stores for tile s done
#endif
    __syncthreads();                        // all waves' tile-s data visible
    if (s + 1 < total) loadTile(s + 1, cur ^ 1);

    v16bf af[2], bfv[4];
#pragma unroll
    for (int mi = 0; mi < 2; ++mi) {
      // A 16-bit 16x32 layout: lane m = ln; k = hf*8 + j (j<8) and
      // 16 + hf*8 + (j-8)  -> two 16B chunks
      int mr = wM * 32 + mi * 16 + ln;
      af[mi] = make_frag2(&sA[cur][mr * LDK + hf * 8],
                          &sA[cur][mr * LDK + hf * 8 + 16]);
    }
#pragma unroll
    for (int ni = 0; ni < 4; ++ni) {
      // B 16-bit 32x16 layout: lane n = ln; k = hf*8 + j (j<8) and
      // 16 + hf*8 + (j-8)  -> two 16B chunks
      int nr = wN * 64 + ni * 16 + ln;
      bfv[ni] = make_frag2(&sB[cur][nr * LDK + hf * 8],
                           &sB[cur][nr * LDK + hf * 8 + 16]);
    }
#pragma unroll
    for (int mi = 0; mi < 2; ++mi)
#pragma unroll
      for (int ni = 0; ni < 4; ++ni)
        c[mi][ni] = __builtin_amdgcn_wmma_f32_16x16x32_bf16(
            false, af[mi], false, bfv[ni], (short)0, c[mi][ni],
            false, false);
  }

  // ---- epilogue ----
  // C layout: VGPR g -> M = g + hf*8 ; N = ln
  if (ACC_R) {
#pragma unroll
    for (int ni = 0; ni < 4; ++ni) {
      int n = n0 + wN * 64 + ni * 16 + ln;
      float bn = bias[n];
#pragma unroll
      for (int mi = 0; mi < 2; ++mi)
#pragma unroll
        for (int g = 0; g < 8; ++g) {
          int m = m0 + wM * 32 + mi * 16 + hf * 8 + g;
          float v = c[mi][ni][g] + bn;
          outF[(size_t)m * N + n] = (v >= 0.f) ? v : 0.1f * v;
        }
    }
  } else {
    int r = blockIdx.z;
#pragma unroll
    for (int ni = 0; ni < 4; ++ni) {
      int n = n0 + wN * 64 + ni * 16 + ln;
#pragma unroll
      for (int mi = 0; mi < 2; ++mi) {
        int mBase = m0 + wM * 32 + mi * 16 + hf * 8;   // 8 consecutive m
        uint4 u;
        u.x = pk_bf16(c[mi][ni][0], c[mi][ni][1]);
        u.y = pk_bf16(c[mi][ni][2], c[mi][ni][3]);
        u.z = pk_bf16(c[mi][ni][4], c[mi][ni][5]);
        u.w = pk_bf16(c[mi][ni][6], c[mi][ni][7]);
        *(uint4*)&outT[((size_t)r * N + n) * M + mBase] = u;
      }
    }
  }
}

// ---------------------------------------------------------------------------
// Launcher
// ---------------------------------------------------------------------------
extern "C" void kernel_launch(void* const* d_in, const int* in_sizes, int n_in,
                              void* d_out, int out_size, void* d_ws,
                              size_t ws_size, hipStream_t stream) {
  (void)n_in; (void)out_size; (void)ws_size;
  const float* p_feat   = (const float*)d_in[0];
  const float* d_feat   = (const float*)d_in[1];
  const float* support  = (const float*)d_in[4];
  const float* p_weight = (const float*)d_in[5];
  const float* p_bias   = (const float*)d_in[6];

  const int P = in_sizes[2];            // 2048
  const int D = in_sizes[3];            // 4096
  const int H = in_sizes[6];            // 256
  const int I = in_sizes[0] / P;        // 512
  const int R = in_sizes[5] / (I * H);  // 5

  char* w = (char*)d_ws;
  auto alloc = [&](size_t bytes) {
    void* p = (void*)w;
    w += (bytes + 255) & ~(size_t)255;
    return p;
  };
  float* rs      = (float*)alloc((size_t)R * P * 4);
  float* colsum  = (float*)alloc((size_t)R * D * 4);
  float* cs      = (float*)alloc((size_t)R * D * 4);
  unsigned short* wcumT = (unsigned short*)alloc((size_t)R * H * I * 2);
  unsigned short* tmpTp = (unsigned short*)alloc((size_t)R * H * P * 2);
  unsigned short* tmpTd = (unsigned short*)alloc((size_t)R * H * D * 2);

  // atomic accumulator must be zeroed every call (graph-capture-safe)
  (void)hipMemsetAsync(colsum, 0, (size_t)R * D * 4, stream);

  row_scale_kernel<<<(R * P + 7) / 8, 256, 0, stream>>>(support, rs, R * P, D);
  col_sum_kernel<<<dim3(R, D / 256, P / 128), 256, 0, stream>>>(support, colsum, P, D);
  inv_sqrt_kernel<<<(R * D + 255) / 256, 256, 0, stream>>>(colsum, cs, R * D);
  wcum_kernel<<<(I * H + 255) / 256, 256, 0, stream>>>(p_weight, wcumT, I, H, R);

  // tmp_p^T[r][h][p] and tmp_d^T[r][h][d]  (bf16, K-contiguous for big GEMMs)
  gemm_bf16<0, 0, 0><<<dim3(P / BMT, H / BNT, R), 256, 0, stream>>>(
      p_feat, wcumT, nullptr, nullptr, nullptr, nullptr, tmpTp, P, H, I, R);
  gemm_bf16<0, 0, 0><<<dim3(D / BMT, H / BNT, R), 256, 0, stream>>>(
      d_feat, wcumT, nullptr, nullptr, nullptr, nullptr, tmpTd, D, H, I, R);

  float* z_p = (float*)d_out;
  float* z_d = z_p + (size_t)P * H;
  // z_p[p][h] = leaky( sum_r norm[r] @ tmp_d[r] + bias )
  gemm_bf16<0, 1, 1><<<dim3(P / BMT, H / BNT, 1), 256, 0, stream>>>(
      support, tmpTd, rs, cs, p_bias, z_p, nullptr, P, H, D, R);
  // z_d[d][h] = leaky( sum_r norm[r]^T @ tmp_p[r] + bias )
  gemm_bf16<1, 1, 1><<<dim3(D / BMT, H / BNT, 1), 256, 0, stream>>>(
      support, tmpTp, cs, rs, p_bias, z_d, nullptr, D, H, P, R);
}
